// StochasticTwoLayerRGAT_attention_35648228557577
// MI455X (gfx1250) — matrix operand order repaired and checked by reading, not compiled, not run on hardware
//
#include <hip/hip_runtime.h>
#include <hip/hip_fp16.h>

#define N_NODES 20000
#define N_EDGES 320000
#define R_REL   3
#define F_DIM   128
#define HEADS   2
#define D_HEAD  64
#define ATTH    32
#define NEG_SLOPE 0.2f

typedef __attribute__((ext_vector_type(16))) _Float16 v16h;
typedef __attribute__((ext_vector_type(8)))  _Float16 v8h;
typedef __attribute__((ext_vector_type(8)))  float    v8f;

// ---------- helpers ----------
__device__ __forceinline__ unsigned fenc(float x) {
  unsigned u = __float_as_uint(x);
  return (u & 0x80000000u) ? ~u : (u | 0x80000000u);
}
__device__ __forceinline__ float fdec(unsigned u) {
  unsigned b = (u & 0x80000000u) ? (u & 0x7FFFFFFFu) : ~u;
  return __uint_as_float(b);
}

// ---------- kernels ----------
__global__ void cast_f32_to_f16(const float* __restrict__ src, _Float16* __restrict__ dst, int n) {
  int i = blockIdx.x * blockDim.x + threadIdx.x;
  if (i < n) dst[i] = (_Float16)src[i];
}

__global__ void fill_u32(unsigned* __restrict__ p, unsigned v, int n) {
  int i = blockIdx.x * blockDim.x + threadIdx.x;
  if (i < n) p[i] = v;
}

// Pre-pack W[r][k][n] (f32, row-major) into the WMMA B-fragment layout (f16):
// Wp[(((r*8 + nt)*4 + ks)*32 + lane)*16 + e] = (f16) W[r][ks*32 + 16*(lane>>4) + e][nt*16 + (lane&15)]
// so each lane's per-K-step B fragment is 16 contiguous halfs (32 B).
__global__ void pack_w_frag(const float* __restrict__ W, _Float16* __restrict__ Wp) {
  int i = blockIdx.x * blockDim.x + threadIdx.x;
  if (i >= R_REL * F_DIM * F_DIM) return;
  int e    = i & 15;
  int lane = (i >> 4) & 31;
  int ks   = (i >> 9) & 3;
  int nt   = (i >> 11) & 7;
  int r    = i >> 14;
  int g = lane >> 4, n = lane & 15;
  int k = ks * 32 + 16 * g + e;
  Wp[i] = (_Float16)W[((size_t)r * F_DIM + k) * F_DIM + nt * 16 + n];
}

// One wave (32 lanes) computes a 16x16 f32 tile of F[r] = A @ W[r].
// grid = (1250 m-tiles, 8 n-tiles, 3 relations), block = 32.
__global__ void gemm_wmma_f16(const _Float16* __restrict__ A,   // [N,128] f16 activations
                              const _Float16* __restrict__ Wp,  // packed B fragments
                              float* __restrict__ F)            // [R,N,128] f32 out
{
  const int lane = threadIdx.x;
  const int mt = blockIdx.x, nt = blockIdx.y, r = blockIdx.z;
  const int g  = lane >> 4;        // lane group (K split)
  const int mn = lane & 15;        // row (A) / col (B,C,D)

  const _Float16* arow = A + (size_t)(mt * 16 + mn) * F_DIM;
  const _Float16* bbase = Wp + ((size_t)(r * 8 + nt) * 4 * 32 + lane) * 16;

  v8f c = {};
#pragma unroll
  for (int ks = 0; ks < 4; ++ks) {
    const int k0 = ks * 32;
    if (ks < 3) {
      __builtin_prefetch(arow + k0 + 32, 0, 1);
      __builtin_prefetch(bbase + (size_t)(ks + 1) * 32 * 16, 0, 1);
    }
    // A 16x32 f16 fragment: elements 0..15 = arow[k0+16g .. +15]
    v8h a0 = *(const v8h*)(arow + k0 + 16 * g);
    v8h a1 = *(const v8h*)(arow + k0 + 16 * g + 8);
    v16h a;
#pragma unroll
    for (int e = 0; e < 8; ++e) { a[e] = a0[e]; a[8 + e] = a1[e]; }
    // B 32x16 f16 fragment: pre-packed contiguous 16 halfs per lane
    v16h b = *(const v16h*)(bbase + (size_t)ks * 32 * 16);
    c = __builtin_amdgcn_wmma_f32_16x16x32_f16(false, a, false, b, (short)0, c, false, false);
  }
  // D 16x16 f32: VGPR j -> M = j + 8g, N = lane&15
  float* out = F + ((size_t)r * N_NODES + (size_t)mt * 16) * F_DIM + nt * 16;
#pragma unroll
  for (int j = 0; j < 8; ++j)
    out[(size_t)(j + 8 * g) * F_DIM + mn] = c[j];
}

// el[r,n,h] = sum_d f[r,n,h,d]*al[r,h,d] ; er analogous. thread per (r,n,h).
__global__ void node_attn_proj(const float* __restrict__ F, const float* __restrict__ al,
                               const float* __restrict__ ar, float* __restrict__ el,
                               float* __restrict__ er) {
  int idx = blockIdx.x * blockDim.x + threadIdx.x;
  if (idx >= R_REL * N_NODES * HEADS) return;
  int h  = idx & 1;
  int rn = idx >> 1;
  int r  = rn / N_NODES;
  const float* f   = F + (size_t)rn * F_DIM + h * D_HEAD;
  const float* alp = al + (r * HEADS + h) * D_HEAD;
  const float* arp = ar + (r * HEADS + h) * D_HEAD;
  float sl = 0.f, sr = 0.f;
#pragma unroll 4
  for (int d = 0; d < D_HEAD; ++d) { float v = f[d]; sl += v * alp[d]; sr += v * arp[d]; }
  el[idx] = sl; er[idx] = sr;
}

// pass 1: segment max per destination (order-preserving uint atomicMax)
__global__ void edge_max_kernel(const int* __restrict__ edges, const float* __restrict__ el,
                                const float* __restrict__ er, unsigned* __restrict__ mx) {
  int i = blockIdx.x * blockDim.x + threadIdx.x;
  if (i >= R_REL * N_EDGES) return;
  int r = i / N_EDGES, e = i - r * N_EDGES;
  int src = edges[(r * 2 + 0) * N_EDGES + e];
  int dst = edges[(r * 2 + 1) * N_EDGES + e];
#pragma unroll
  for (int h = 0; h < HEADS; ++h) {
    float v = el[(r * N_NODES + src) * HEADS + h] + er[(r * N_NODES + dst) * HEADS + h];
    v = v > 0.f ? v : NEG_SLOPE * v;
    atomicMax(&mx[(r * N_NODES + dst) * HEADS + h], fenc(v));
  }
}

// pass 2: ex = exp(e - m[dst]); den[dst] += ex
__global__ void edge_expsum_kernel(const int* __restrict__ edges, const float* __restrict__ el,
                                   const float* __restrict__ er, const unsigned* __restrict__ mx,
                                   float* __restrict__ exb, float* __restrict__ den) {
  int i = blockIdx.x * blockDim.x + threadIdx.x;
  if (i >= R_REL * N_EDGES) return;
  int r = i / N_EDGES, e = i - r * N_EDGES;
  int src = edges[(r * 2 + 0) * N_EDGES + e];
  int dst = edges[(r * 2 + 1) * N_EDGES + e];
#pragma unroll
  for (int h = 0; h < HEADS; ++h) {
    float v = el[(r * N_NODES + src) * HEADS + h] + er[(r * N_NODES + dst) * HEADS + h];
    v = v > 0.f ? v : NEG_SLOPE * v;
    float m  = fdec(mx[(r * N_NODES + dst) * HEADS + h]);
    float ev = __expf(v - m);
    exb[((size_t)r * N_EDGES + e) * HEADS + h] = ev;
    atomicAdd(&den[(r * N_NODES + dst) * HEADS + h], ev);
  }
}

// pass 3: Z[dst, r, :] += alpha * F[r, src, :]. block = 128 threads = one edge.
__global__ void edge_aggregate_kernel(const int* __restrict__ edges, const float* __restrict__ exb,
                                      const float* __restrict__ den, const float* __restrict__ F,
                                      float* __restrict__ Z) {
  int e = blockIdx.x, r = blockIdx.y, d = threadIdx.x;
  int src = edges[(r * 2 + 0) * N_EDGES + e];
  int dst = edges[(r * 2 + 1) * N_EDGES + e];
  int h = d >> 6;
  float a = exb[((size_t)r * N_EDGES + e) * HEADS + h] /
            fmaxf(den[(r * N_NODES + dst) * HEADS + h], 1e-9f);
  float v = a * F[((size_t)r * N_NODES + src) * F_DIM + d];
  atomicAdd(&Z[((size_t)dst * R_REL + r) * F_DIM + d], v);
}

// bias (+ optional relu) in-place on Z[n,r,:], then semantic score
// w[n,r] = tanh(z @ w1 + bb) @ w2 ; wsum[r] += w[n,r]/N.  block = 32 (one wave), grid=(N,R).
__global__ void finalize_score_kernel(float* __restrict__ Z, const float* __restrict__ bias,
                                      const float* __restrict__ w1, const float* __restrict__ bb,
                                      const float* __restrict__ w2, float* __restrict__ wsum,
                                      int do_relu) {
  int n = blockIdx.x, r = blockIdx.y, lane = threadIdx.x;
  float* zr = Z + ((size_t)n * R_REL + r) * F_DIM;
  for (int t = lane; t < F_DIM; t += 32) {
    float v = zr[t] + bias[r * F_DIM + t];
    if (do_relu) v = fmaxf(v, 0.f);
    zr[t] = v;
  }
  __syncthreads();
  float acc = bb[lane];                 // lane == attention hidden dim (ATTH==32)
  for (int k = 0; k < F_DIM; ++k) acc += zr[k] * w1[k * ATTH + lane];
  float p = tanhf(acc) * w2[lane];
  for (int off = 16; off; off >>= 1) p += __shfl_down(p, off, 32);
  if (lane == 0) atomicAdd(&wsum[r], p * (1.0f / N_NODES));
}

__global__ void softmax3_kernel(const float* __restrict__ wsum, float* __restrict__ beta) {
  if (threadIdx.x == 0 && blockIdx.x == 0) {
    float a = wsum[0], b = wsum[1], c = wsum[2];
    float m = fmaxf(a, fmaxf(b, c));
    float ea = __expf(a - m), eb = __expf(b - m), ec = __expf(c - m);
    float s = ea + eb + ec;
    beta[0] = ea / s; beta[1] = eb / s; beta[2] = ec / s;
  }
}

__global__ void combine_kernel(const float* __restrict__ Z, const float* __restrict__ beta,
                               float* __restrict__ out) {
  int i = blockIdx.x * blockDim.x + threadIdx.x;
  if (i >= N_NODES * F_DIM) return;
  int n = i / F_DIM, d = i - n * F_DIM;
  const float* z = Z + (size_t)n * R_REL * F_DIM + d;
  out[i] = beta[0] * z[0] + beta[1] * z[F_DIM] + beta[2] * z[2 * F_DIM];
}

// ---------- host ----------
static void run_layer(const float* act, const int* edges, const float* W, const float* al,
                      const float* ar, const float* bias, const float* attw1, const float* attb,
                      const float* attw2, float* dest, int do_relu,
                      _Float16* hf16, _Float16* wpack, float* F, float* el, float* er,
                      unsigned* mx, float* den, float* exb, float* Z, float* wsum, float* beta,
                      hipStream_t stream) {
  const int nact = N_NODES * F_DIM;
  const int nw   = R_REL * F_DIM * F_DIM;
  const int nnh  = R_REL * N_NODES * HEADS;
  const int nre  = R_REL * N_EDGES;

  cast_f32_to_f16<<<(nact + 255) / 256, 256, 0, stream>>>(act, hf16, nact);
  pack_w_frag<<<(nw + 255) / 256, 256, 0, stream>>>(W, wpack);

  hipMemsetAsync(Z, 0, (size_t)N_NODES * R_REL * F_DIM * sizeof(float), stream);
  hipMemsetAsync(den, 0, (size_t)nnh * sizeof(float), stream);
  hipMemsetAsync(wsum, 0, 2 * 16 * sizeof(float), stream);   // wsum + beta region
  fill_u32<<<(nnh + 255) / 256, 256, 0, stream>>>(mx, 0x007FFFFFu /* enc(-inf) */, nnh);

  gemm_wmma_f16<<<dim3(N_NODES / 16, F_DIM / 16, R_REL), 32, 0, stream>>>(hf16, wpack, F);
  node_attn_proj<<<(nnh + 255) / 256, 256, 0, stream>>>(F, al, ar, el, er);
  edge_max_kernel<<<(nre + 255) / 256, 256, 0, stream>>>(edges, el, er, mx);
  edge_expsum_kernel<<<(nre + 255) / 256, 256, 0, stream>>>(edges, el, er, mx, exb, den);
  edge_aggregate_kernel<<<dim3(N_EDGES, R_REL), F_DIM, 0, stream>>>(edges, exb, den, F, Z);
  finalize_score_kernel<<<dim3(N_NODES, R_REL), ATTH, 0, stream>>>(Z, bias, attw1, attb, attw2,
                                                                   wsum, do_relu);
  softmax3_kernel<<<1, 1, 0, stream>>>(wsum, beta);
  combine_kernel<<<(nact + 255) / 256, 256, 0, stream>>>(Z, beta, dest);
}

extern "C" void kernel_launch(void* const* d_in, const int* in_sizes, int n_in,
                              void* d_out, int out_size, void* d_ws, size_t ws_size,
                              hipStream_t stream) {
  (void)in_sizes; (void)n_in; (void)out_size;
  const float* x    = (const float*)d_in[0];
  const int*   edg  = (const int*)d_in[1];
  const float* W1   = (const float*)d_in[2];
  const float* al1  = (const float*)d_in[3];
  const float* ar1  = (const float*)d_in[4];
  const float* b1   = (const float*)d_in[5];
  const float* W2   = (const float*)d_in[6];
  const float* al2  = (const float*)d_in[7];
  const float* ar2  = (const float*)d_in[8];
  const float* b2   = (const float*)d_in[9];
  const float* aw1  = (const float*)d_in[10];
  const float* ab1  = (const float*)d_in[11];
  const float* aw2  = (const float*)d_in[12];
  const float* bw1  = (const float*)d_in[13];
  const float* bb1  = (const float*)d_in[14];
  const float* bw2  = (const float*)d_in[15];
  float* out = (float*)d_out;

  auto al256 = [](size_t x) { return (x + 255) & ~(size_t)255; };
  size_t o = 0;
  size_t off_hf16 = o; o += al256((size_t)N_NODES * F_DIM * 2);
  size_t off_wp   = o; o += al256((size_t)R_REL * F_DIM * F_DIM * 2);
  size_t off_f    = o; o += al256((size_t)R_REL * N_NODES * F_DIM * 4);
  size_t off_el   = o; o += al256((size_t)R_REL * N_NODES * HEADS * 4);
  size_t off_er   = o; o += al256((size_t)R_REL * N_NODES * HEADS * 4);
  size_t off_mx   = o; o += al256((size_t)R_REL * N_NODES * HEADS * 4);
  size_t off_den  = o; o += al256((size_t)R_REL * N_NODES * HEADS * 4);
  size_t off_ex   = o; o += al256((size_t)R_REL * N_EDGES * HEADS * 4);
  size_t off_z    = o; o += al256((size_t)N_NODES * R_REL * F_DIM * 4);
  size_t off_hmid = o; o += al256((size_t)N_NODES * F_DIM * 4);
  size_t off_sm   = o; o += 256;                    // wsum[3] @ +0, beta[3] @ +64
  if (o > ws_size) return;                          // scratch too small: bail

  char* ws = (char*)d_ws;
  _Float16* hf16  = (_Float16*)(ws + off_hf16);
  _Float16* wpack = (_Float16*)(ws + off_wp);
  float*    F     = (float*)(ws + off_f);
  float*    el    = (float*)(ws + off_el);
  float*    er    = (float*)(ws + off_er);
  unsigned* mx    = (unsigned*)(ws + off_mx);
  float*    den   = (float*)(ws + off_den);
  float*    exb   = (float*)(ws + off_ex);
  float*    Z     = (float*)(ws + off_z);
  float*    hmid  = (float*)(ws + off_hmid);
  float*    wsum  = (float*)(ws + off_sm);
  float*    beta  = (float*)(ws + off_sm + 64);

  // layer 1: GAT per relation -> relu -> semantic attention -> hmid
  run_layer(x, edg, W1, al1, ar1, b1, aw1, ab1, aw2, hmid, /*relu=*/1,
            hf16, wpack, F, el, er, mx, den, exb, Z, wsum, beta, stream);
  // layer 2: GAT per relation -> semantic attention -> out
  run_layer(hmid, edg, W2, al2, ar2, b2, bw1, bb1, bw2, out, /*relu=*/0,
            hf16, wpack, F, el, er, mx, den, exb, Z, wsum, beta, stream);
}
